// ProbSparseAttention_40037685133470
// MI455X (gfx1250) — matrix operand order repaired and checked by reading, not compiled, not run on hardware
//
#include <hip/hip_runtime.h>
#include <hip/hip_bf16.h>
#include <math.h>

// Problem constants (from reference)
#define BB 16
#define HH 8
#define LL 2048
#define DD 64
#define UU 40          // U = 5 * ceil(ln(2048)) = 40
#define UP 48          // U padded to 3x16 row tiles
#define BHN 128        // B*H
#define NCH 8          // L-chunks per (b,h)
#define CHK 256        // rows per chunk
#define LT 32          // l-tile (K-dim of the P*V WMMA)

typedef __attribute__((ext_vector_type(16))) __bf16 v16bf;
typedef __attribute__((ext_vector_type(8)))  float  v8f;
typedef __attribute__((ext_vector_type(8)))  unsigned short us8;
typedef __attribute__((ext_vector_type(4)))  float  f4v;
typedef __attribute__((ext_vector_type(4)))  unsigned int ui4;
typedef __attribute__((ext_vector_type(8)))  int i8v;
typedef __attribute__((ext_vector_type(4)))  int i4v;

#if defined(__has_builtin)
# if __has_builtin(__builtin_amdgcn_tensor_load_to_lds) && __has_builtin(__builtin_amdgcn_s_wait_tensorcnt)
#  define HAVE_TDM 1
# endif
#endif
#ifndef HAVE_TDM
# define HAVE_TDM 0
#endif

union BFrag { v16bf v; us8 h[2]; };

__device__ __forceinline__ unsigned short f2bf(float f) {
  unsigned int u = __float_as_uint(f);
  unsigned int r = (u + 0x7FFFu + ((u >> 16) & 1u)) >> 16;   // RNE
  return (unsigned short)r;
}

// ---------------- workspace layout (bytes) ----------------
static constexpr size_t OFF_IDX  = 0;                                   // 40 i32
static constexpr size_t OFF_MTOP = 256;                                 // 128*40 i32
static constexpr size_t OFF_M    = OFF_MTOP + (size_t)BHN*UU*4;         // 128*2048 f32
static constexpr size_t OFF_PO   = OFF_M   + (size_t)BHN*LL*4;          // 1024*48*64 f32
static constexpr size_t OFF_PM   = OFF_PO  + (size_t)BHN*NCH*UP*DD*4;   // 1024*48 f32
static constexpr size_t OFF_PS   = OFF_PM  + (size_t)BHN*NCH*UP*4;      // 1024*48 f32
static constexpr size_t OFF_PV   = OFF_PS  + (size_t)BHN*NCH*UP*4;      // 1024*64 f32
static constexpr size_t OFF_UPD  = OFF_PV  + (size_t)BHN*NCH*DD*4;      // 128*48*64 f32
static constexpr size_t OFF_VS   = OFF_UPD + (size_t)BHN*UP*DD*4;       // 128*64 f32
// total ~15.9 MB

// ---------------- kernel 1: threefry sampling indices ----------------
__device__ __forceinline__ void tf_r(unsigned int& x0, unsigned int& x1, int r) {
  x0 += x1; x1 = (x1 << r) | (x1 >> (32 - r)); x1 ^= x0;
}
__device__ void threefry2x32(unsigned int k0, unsigned int k1,
                             unsigned int c0, unsigned int c1,
                             unsigned int& o0, unsigned int& o1) {
  unsigned int ks2 = k0 ^ k1 ^ 0x1BD11BDAu;
  unsigned int x0 = c0 + k0, x1 = c1 + k1;
  tf_r(x0,x1,13); tf_r(x0,x1,15); tf_r(x0,x1,26); tf_r(x0,x1,6);
  x0 += k1;  x1 += ks2 + 1u;
  tf_r(x0,x1,17); tf_r(x0,x1,29); tf_r(x0,x1,16); tf_r(x0,x1,24);
  x0 += ks2; x1 += k0 + 2u;
  tf_r(x0,x1,13); tf_r(x0,x1,15); tf_r(x0,x1,26); tf_r(x0,x1,6);
  x0 += k0;  x1 += k1 + 3u;
  tf_r(x0,x1,17); tf_r(x0,x1,29); tf_r(x0,x1,16); tf_r(x0,x1,24);
  x0 += k1;  x1 += ks2 + 4u;
  tf_r(x0,x1,13); tf_r(x0,x1,15); tf_r(x0,x1,26); tf_r(x0,x1,6);
  x0 += ks2; x1 += k0 + 5u;
  o0 = x0; o1 = x1;
}

__global__ void k_rng(int* __restrict__ idx) {
  int t = threadIdx.x;
  if (t >= 20) return;
  // key(42) = [0,42]; split -> take second derived key (x1 outputs of counters (0,2),(1,3))
  unsigned int a0, a1, b0, b1;
  threefry2x32(0u, 42u, 0u, 2u, a0, a1);
  threefry2x32(0u, 42u, 1u, 3u, b0, b1);
  unsigned int o0, o1;
  threefry2x32(a1, b1, (unsigned)t, (unsigned)(t + 20), o0, o1);
  idx[t]      = (int)(o0 & (LL - 1));   // 2^32 % 2048 == 0 -> plain mod is unbiased
  idx[t + 20] = (int)(o1 & (LL - 1));
}

// ---------------- kernel 2: M = rowmax(Q K_s^T) - rowsum/L ----------------
__global__ __launch_bounds__(256) void k_scoreM(const float* __restrict__ q,
                                                const float* __restrict__ k,
                                                const int* __restrict__ idx,
                                                float* __restrict__ Mout) {
  __shared__ unsigned short Ks[UP][DD];       // gathered K_sample, bf16, zero-padded
  __shared__ unsigned short Qt[8][16][DD];    // per-wave Q tile
  int tid = threadIdx.x;
  int blk = blockIdx.x;                       // bh*8 + chunk
  int bh = blk >> 3, ch = blk & 7;
  size_t gbase = (size_t)bh * LL * DD;

  for (int s = tid; s < UP * 16; s += 256) {  // 48 rows x 16 float4
    int r = s >> 4, q4 = s & 15;
    unsigned short h0 = 0, h1 = 0, h2 = 0, h3 = 0;
    if (r < UU) {
      float4 kv = *(const float4*)(k + gbase + (size_t)idx[r] * DD + q4 * 4);
      h0 = f2bf(kv.x); h1 = f2bf(kv.y); h2 = f2bf(kv.z); h3 = f2bf(kv.w);
    }
    Ks[r][q4*4+0] = h0; Ks[r][q4*4+1] = h1; Ks[r][q4*4+2] = h2; Ks[r][q4*4+3] = h3;
  }
  __syncthreads();

  int w = tid >> 5, lane = tid & 31;
  int n = lane & 15, hi = lane >> 4;
  for (int t = 0; t < 2; ++t) {
    int lb = ch * CHK + (w * 2 + t) * 16;
    for (int s = lane; s < 16 * 16; s += 32) {          // stage this wave's Q tile
      int r = s >> 4, q4 = s & 15;
      float4 qv = *(const float4*)(q + gbase + (size_t)(lb + r) * DD + q4 * 4);
      Qt[w][r][q4*4+0] = f2bf(qv.x); Qt[w][r][q4*4+1] = f2bf(qv.y);
      Qt[w][r][q4*4+2] = f2bf(qv.z); Qt[w][r][q4*4+3] = f2bf(qv.w);
    }
    float rmax[8], rsum[8];
    for (int i = 0; i < 8; ++i) { rmax[i] = -1e30f; rsum[i] = 0.f; }
    for (int nt = 0; nt < 3; ++nt) {
      v8f c = {};
      for (int ec = 0; ec < 2; ++ec) {
        BFrag a, b;
        const unsigned short* ap = &Qt[w][n][ec * 32 + hi * 8];
        a.h[0] = *(const us8*)ap;        a.h[1] = *(const us8*)(ap + 16);
        const unsigned short* bp = &Ks[nt * 16 + n][ec * 32 + hi * 8];
        b.h[0] = *(const us8*)bp;        b.h[1] = *(const us8*)(bp + 16);
        c = __builtin_amdgcn_wmma_f32_16x16x32_bf16(false, a.v, false, b.v,
                                                    (short)0, c, false, false);
      }
      int col = nt * 16 + n;
      bool ok = col < UU;
      for (int i = 0; i < 8; ++i) {
        float vmx = ok ? c[i] : -1e30f;
        float vsm = ok ? c[i] : 0.f;
        for (int m = 1; m < 16; m <<= 1) {
          vmx = fmaxf(vmx, __shfl_xor(vmx, m, 32));
          vsm += __shfl_xor(vsm, m, 32);
        }
        rmax[i] = fmaxf(rmax[i], vmx);
        rsum[i] += vsm;
      }
    }
    if (n == 0) {
      for (int i = 0; i < 8; ++i) {
        int row = i + hi * 8;
        Mout[(size_t)bh * LL + lb + row] = rmax[i] - rsum[i] * (1.0f / (float)LL);
      }
    }
  }
}

// ---------------- kernel 3: top-40 per (b,h) ----------------
__global__ __launch_bounds__(256) void k_topk(const float* __restrict__ Mv,
                                              int* __restrict__ mtop) {
  __shared__ float sm[LL];
  __shared__ float rv[256];
  __shared__ int   ri[256];
  int bh = blockIdx.x, t = threadIdx.x;

#if HAVE_TDM
  // Tensor Data Mover: DMA the 2048-float M row into LDS (one 1-row 2D tile).
  if (t < 32) {   // wave 0 only; TDM ignores EXEC, issued once per wave
    unsigned int lds_off = (unsigned int)(unsigned long long)
                           (__attribute__((address_space(3))) float*)sm;
    unsigned long long ga = (unsigned long long)(Mv + (size_t)bh * LL);
    ui4 g0 = { 1u,                                  // count=1 (user descriptor)
               lds_off,                             // lds_addr
               (unsigned int)ga,                    // global_addr[31:0]
               (unsigned int)(((ga >> 32) & 0x1FFFFFFull) | (2u << 30)) }; // addr[56:32] | type=2
    i8v g1 = { 0x20000,                 // data_size=2 (4 bytes), no mask/flags
               (int)(2048u << 16),      // tensor_dim0 low16 in bits[63:48]
               0x10000,                 // tensor_dim0 hi16=0 | tensor_dim1=1
               (int)(2048u << 16),      // tensor_dim1 hi16=0 | tile_dim0=2048
               1,                       // tile_dim1=1 (tile_dim2=0)
               2048,                    // tensor_dim0_stride low32
               0, 0 };
    i4v gz = { 0, 0, 0, 0 };
    i8v gz8 = { 0, 0, 0, 0, 0, 0, 0, 0 };
    __builtin_amdgcn_tensor_load_to_lds(g0, g1, gz, gz, gz8, 0);  // 6-arg toolchain form
    __builtin_amdgcn_s_wait_tensorcnt(0);
  }
  __syncthreads();
#else
  for (int i = t; i < LL; i += 256) sm[i] = Mv[(size_t)bh * LL + i];
  __syncthreads();
#endif

  for (int it = 0; it < UU; ++it) {
    float bv = -1e30f; int bi = 0;
    for (int i = t; i < LL; i += 256) {
      float v = sm[i];
      if (v > bv) { bv = v; bi = i; }          // ascending scan: lowest index wins ties
    }
    rv[t] = bv; ri[t] = bi;
    __syncthreads();
    for (int s = 128; s > 0; s >>= 1) {
      if (t < s) {
        float v2 = rv[t + s]; int i2 = ri[t + s];
        if (v2 > rv[t] || (v2 == rv[t] && i2 < ri[t])) { rv[t] = v2; ri[t] = i2; }
      }
      __syncthreads();
    }
    if (t == 0) { mtop[bh * UU + it] = ri[0]; sm[ri[0]] = -1e30f; }
    __syncthreads();
  }
}

// ---------------- kernel 4: flash attention over 48 padded rows ----------------
__global__ __launch_bounds__(192) void k_flash(const float* __restrict__ q,
                                               const float* __restrict__ k,
                                               const float* __restrict__ v,
                                               const int* __restrict__ mtop,
                                               float* __restrict__ pO,
                                               float* __restrict__ pM,
                                               float* __restrict__ pS,
                                               float* __restrict__ pV) {
  __shared__ unsigned short Qr[UP][DD];        // gathered Q rows (bf16)
  __shared__ unsigned short Kt[LT][DD];        // K tile
  __shared__ unsigned short VT[DD][LT + 8];    // V tile transposed [e][l], padded
  __shared__ unsigned short Pt[UP][LT];        // exp(S - m)
  __shared__ float m_run[UP], s_run[UP], m_new[UP], alpha[UP];
  __shared__ float fmax2[UP][2], fsum2[UP][2];
  __shared__ float vsum[DD];

  int tid = threadIdx.x;
  int blk = blockIdx.x;                        // bh*8 + chunk
  int bh = blk >> 3, ch = blk & 7;
  size_t gbase = (size_t)bh * LL * DD;

  if (tid < UP) { m_run[tid] = -1e30f; s_run[tid] = 0.f; }
  if (tid < DD) vsum[tid] = 0.f;

  for (int s = tid; s < UP * 16; s += 192) {   // gather Q_reduce, pad rows >= 40
    int r = s >> 4, q4 = s & 15;
    unsigned short h0 = 0, h1 = 0, h2 = 0, h3 = 0;
    if (r < UU) {
      int src = mtop[bh * UU + r];
      float4 qv = *(const float4*)(q + gbase + (size_t)src * DD + q4 * 4);
      h0 = f2bf(qv.x); h1 = f2bf(qv.y); h2 = f2bf(qv.z); h3 = f2bf(qv.w);
    }
    Qr[r][q4*4+0] = h0; Qr[r][q4*4+1] = h1; Qr[r][q4*4+2] = h2; Qr[r][q4*4+3] = h3;
  }
  __syncthreads();

  int lane = tid & 31, w = tid >> 5;
  int n = lane & 15, hi = lane >> 4;
  int su = w >> 1, slt = w & 1;      // S fragment: u-tile, l-tile
  int ou = su, oe0 = (w & 1) * 2;    // O fragments: u-tile, e-tiles oe0, oe0+1

  v8f o0 = {}, o1 = {};
  float4 vacc = make_float4(0.f, 0.f, 0.f, 0.f);
  int myq4 = tid & 15;

  for (int step = 0; step < CHK / LT; ++step) {
    int l0 = ch * CHK + step * LT;
    for (int s = tid; s < LT * 16; s += 192) {            // stage K, V (transposed)
      int r = s >> 4, q4 = s & 15;
      float4 kv = *(const float4*)(k + gbase + (size_t)(l0 + r) * DD + q4 * 4);
      Kt[r][q4*4+0] = f2bf(kv.x); Kt[r][q4*4+1] = f2bf(kv.y);
      Kt[r][q4*4+2] = f2bf(kv.z); Kt[r][q4*4+3] = f2bf(kv.w);
      float4 vv = *(const float4*)(v + gbase + (size_t)(l0 + r) * DD + q4 * 4);
      VT[q4*4+0][r] = f2bf(vv.x); VT[q4*4+1][r] = f2bf(vv.y);
      VT[q4*4+2][r] = f2bf(vv.z); VT[q4*4+3][r] = f2bf(vv.w);
      vacc.x += vv.x; vacc.y += vv.y; vacc.z += vv.z; vacc.w += vv.w;
      if (step + 1 < CHK / LT) {                          // gfx1250 global_prefetch_b8
        __builtin_prefetch(k + gbase + (size_t)(l0 + LT + r) * DD + q4 * 4, 0, 0);
        __builtin_prefetch(v + gbase + (size_t)(l0 + LT + r) * DD + q4 * 4, 0, 0);
      }
    }
    __syncthreads();

    // S fragment = Qr[u-tile] x Kt^T[l-tile]
    v8f c = {};
    for (int ec = 0; ec < 2; ++ec) {
      BFrag a, b;
      const unsigned short* ap = &Qr[su * 16 + n][ec * 32 + hi * 8];
      a.h[0] = *(const us8*)ap;   a.h[1] = *(const us8*)(ap + 16);
      const unsigned short* bp = &Kt[slt * 16 + n][ec * 32 + hi * 8];
      b.h[0] = *(const us8*)bp;   b.h[1] = *(const us8*)(bp + 16);
      c = __builtin_amdgcn_wmma_f32_16x16x32_bf16(false, a.v, false, b.v,
                                                  (short)0, c, false, false);
    }
    for (int i = 0; i < 8; ++i) {                         // fragment row max
      float vmx = c[i];
      for (int m = 1; m < 16; m <<= 1) vmx = fmaxf(vmx, __shfl_xor(vmx, m, 32));
      if (n == 0) fmax2[su * 16 + i + hi * 8][slt] = vmx;
    }
    __syncthreads();

    if (tid < UP) {                                        // online-softmax stats
      float mo = m_run[tid];
      float mn = fmaxf(mo, fmaxf(fmax2[tid][0], fmax2[tid][1]));
      m_new[tid] = mn;
      alpha[tid] = __expf(mo - mn);
    }
    __syncthreads();

    for (int i = 0; i < 8; ++i) {                          // P = exp(S - m)
      int row = su * 16 + i + hi * 8;
      float p = __expf(c[i] - m_new[row]);
      Pt[row][slt * 16 + n] = f2bf(p);
      float ps = p;
      for (int m = 1; m < 16; m <<= 1) ps += __shfl_xor(ps, m, 32);
      if (n == 0) fsum2[row][slt] = ps;
    }
    __syncthreads();

    if (tid < UP) {
      s_run[tid] = s_run[tid] * alpha[tid] + fsum2[tid][0] + fsum2[tid][1];
      m_run[tid] = m_new[tid];
    }
    {                                                      // rescale O, O += P x V
      float al[8];
      for (int i = 0; i < 8; ++i) al[i] = alpha[ou * 16 + i + hi * 8];
      for (int i = 0; i < 8; ++i) { o0[i] *= al[i]; o1[i] *= al[i]; }
      BFrag pa, vb0, vb1;
      const unsigned short* pp = &Pt[ou * 16 + n][hi * 8];
      pa.h[0] = *(const us8*)pp;   pa.h[1] = *(const us8*)(pp + 16);
      const unsigned short* vp0 = &VT[oe0 * 16 + n][hi * 8];
      vb0.h[0] = *(const us8*)vp0; vb0.h[1] = *(const us8*)(vp0 + 16);
      const unsigned short* vp1 = &VT[(oe0 + 1) * 16 + n][hi * 8];
      vb1.h[0] = *(const us8*)vp1; vb1.h[1] = *(const us8*)(vp1 + 16);
      o0 = __builtin_amdgcn_wmma_f32_16x16x32_bf16(false, pa.v, false, vb0.v,
                                                   (short)0, o0, false, false);
      o1 = __builtin_amdgcn_wmma_f32_16x16x32_bf16(false, pa.v, false, vb1.v,
                                                   (short)0, o1, false, false);
    }
    __syncthreads();
  }

  size_t pbase = (size_t)blk * UP * DD;                    // write partials
  for (int i = 0; i < 8; ++i) {
    int row = ou * 16 + i + hi * 8;
    pO[pbase + (size_t)row * DD + oe0 * 16 + n]       = o0[i];
    pO[pbase + (size_t)row * DD + (oe0 + 1) * 16 + n] = o1[i];
  }
  if (tid < UP) { pM[(size_t)blk * UP + tid] = m_run[tid]; pS[(size_t)blk * UP + tid] = s_run[tid]; }
  atomicAdd(&vsum[myq4 * 4 + 0], vacc.x);
  atomicAdd(&vsum[myq4 * 4 + 1], vacc.y);
  atomicAdd(&vsum[myq4 * 4 + 2], vacc.z);
  atomicAdd(&vsum[myq4 * 4 + 3], vacc.w);
  __syncthreads();
  if (tid < DD) pV[(size_t)blk * DD + tid] = vsum[tid];
}

// ---------------- kernel 5: merge chunk partials ----------------
__global__ __launch_bounds__(256) void k_merge(const float* __restrict__ pO,
                                               const float* __restrict__ pM,
                                               const float* __restrict__ pS,
                                               const float* __restrict__ pV,
                                               float* __restrict__ upd,
                                               float* __restrict__ vs) {
  __shared__ float wgt[UP][NCH];
  int bh = blockIdx.x, t = threadIdx.x;
  if (t < UP) {
    float mx = -1e30f;
    for (int c = 0; c < NCH; ++c) mx = fmaxf(mx, pM[(size_t)(bh * NCH + c) * UP + t]);
    float e_[NCH];
    float ssum = 0.f;
    for (int c = 0; c < NCH; ++c) {
      float e = __expf(pM[(size_t)(bh * NCH + c) * UP + t] - mx);
      e_[c] = e;
      ssum += e * pS[(size_t)(bh * NCH + c) * UP + t];
    }
    float inv = 1.0f / ssum;
    for (int c = 0; c < NCH; ++c) wgt[t][c] = e_[c] * inv;
  }
  __syncthreads();
  for (int s = t; s < UP * DD; s += 256) {
    int r = s / DD, e = s % DD;
    float acc = 0.f;
    for (int c = 0; c < NCH; ++c)
      acc += wgt[r][c] * pO[((size_t)(bh * NCH + c) * UP + r) * DD + e];
    upd[((size_t)bh * UP + r) * DD + e] = acc;
  }
  if (t < DD) {
    float a = 0.f;
    for (int c = 0; c < NCH; ++c) a += pV[(size_t)(bh * NCH + c) * DD + t];
    vs[(size_t)bh * DD + t] = a;
  }
}

// ---------------- kernel 6: broadcast V_sum into context (streaming NT stores) ----------------
__global__ __launch_bounds__(256) void k_bcast(const float* __restrict__ vs,
                                               float* __restrict__ out) {
  size_t i = (size_t)blockIdx.x * 256 + threadIdx.x;   // float4 index
  int bh = (int)(i / ((size_t)LL * 16));
  int e4 = (int)(i & 15);
  f4v vv = *(const f4v*)(vs + (size_t)bh * DD + e4 * 4);
  __builtin_nontemporal_store(vv, (f4v*)(out + i * 4));  // write-once stream: TH=NT
}

// ---------------- kernel 7: scatter updated rows ----------------
__global__ __launch_bounds__(256) void k_scatter(const float* __restrict__ upd,
                                                 const int* __restrict__ mtop,
                                                 float* __restrict__ out) {
  int bh = blockIdx.x, t = threadIdx.x;
  for (int s = t; s < UU * 16; s += 256) {
    int u = s >> 4, q4 = s & 15;
    int l = mtop[bh * UU + u];
    float4 uv = *(const float4*)(upd + ((size_t)bh * UP + u) * DD + q4 * 4);
    *(float4*)(out + ((size_t)bh * LL + l) * DD + q4 * 4) = uv;
  }
}

extern "C" void kernel_launch(void* const* d_in, const int* in_sizes, int n_in,
                              void* d_out, int out_size, void* d_ws, size_t ws_size,
                              hipStream_t stream) {
  const float* q = (const float*)d_in[0];
  const float* k = (const float*)d_in[1];
  const float* v = (const float*)d_in[2];
  float* out = (float*)d_out;
  char* ws = (char*)d_ws;

  int*   idx  = (int*)(ws + OFF_IDX);
  int*   mtop = (int*)(ws + OFF_MTOP);
  float* Mv   = (float*)(ws + OFF_M);
  float* pO   = (float*)(ws + OFF_PO);
  float* pM   = (float*)(ws + OFF_PM);
  float* pS   = (float*)(ws + OFF_PS);
  float* pV   = (float*)(ws + OFF_PV);
  float* upd  = (float*)(ws + OFF_UPD);
  float* vs   = (float*)(ws + OFF_VS);

  k_rng   <<<1,           32, 0, stream>>>(idx);
  k_scoreM<<<BHN * NCH,  256, 0, stream>>>(q, k, idx, Mv);
  k_topk  <<<BHN,        256, 0, stream>>>(Mv, mtop);
  k_flash <<<BHN * NCH,  192, 0, stream>>>(q, k, v, mtop, pO, pM, pS, pV);
  k_merge <<<BHN,        256, 0, stream>>>(pO, pM, pS, pV, upd, vs);
  k_bcast <<<(BHN * LL * 16) / 256, 256, 0, stream>>>(vs, out);
  k_scatter<<<BHN,       256, 0, stream>>>(upd, mtop, out);
}
